// MambaLayer_14937896256045
// MI455X (gfx1250) — matrix operand order repaired and checked by reading, not compiled, not run on hardware
//
#include <hip/hip_runtime.h>

typedef _Float16 half8  __attribute__((ext_vector_type(8)));
typedef _Float16 v16h   __attribute__((ext_vector_type(16)));
typedef float    v8f    __attribute__((ext_vector_type(8)));
typedef unsigned int u32x4 __attribute__((ext_vector_type(4)));
typedef int      i32x4  __attribute__((ext_vector_type(4)));
typedef int      i32x8  __attribute__((ext_vector_type(8)));

#define BATCH   4
#define DMODEL  512
#define SEQ     2048
#define DSTATE  16
#define DCONV   4
#define DINNER  1024
#define DTRANK  32
#define MROWS   (BATCH * SEQ)          // 8192

// ---------------------------------------------------------------------------
// WMMA helpers (CDNA5 16x16x32 f16 -> f32), contiguous-K lane order, same for A and B
// ---------------------------------------------------------------------------
__device__ __forceinline__ v8f wmma_f16(v16h a, v16h b, v8f c) {
  return __builtin_amdgcn_wmma_f32_16x16x32_f16(false, a, false, b,
                                                (short)0, c, false, false);
}

// A tile 16x32 (MxK) f16, row-major, leading dim lda.
// lane<16: M=lane,    K {0..15}
// lane>=16: M=lane-16, K {16..31}
template <typename PT>
__device__ __forceinline__ v16h load_A_tile(PT A, int lda, int lane) {
  int m  = lane & 15;
  int kb = (lane < 16) ? 0 : 16;
  PT p = A + (size_t)m * lda + kb;
  return *(const v16h*)(p);
}

// B tile 32x16 (KxN) f16 supplied as rows of W (B^T rows), W row-major [N][K].
// lane holds column n=lane&15, contiguous K range base (lane<16?0:16)
template <typename PT>
__device__ __forceinline__ v16h load_B_tile(PT W, int ldw, int lane) {
  int n  = lane & 15;
  int kb = (lane < 16) ? 0 : 16;
  return *(const v16h*)(W + (size_t)n * ldw + kb);
}

// ---------------------------------------------------------------------------
// TDM panel stage: copy W[nrows x kdim] f16 row-major panel into LDS.
// Builds a 2D Tensor DMA Descriptor (ISA §8.3/8.4) and issues
// TENSOR_LOAD_TO_LDS from wave 0 only (EXEC ignored by TDM).
// Toolchain here uses the 6-arg builtin: (g0, g1, g2, g3, g_extra, cpol).
// ---------------------------------------------------------------------------
__device__ __forceinline__ void stage_panel(const _Float16* gsrc, _Float16* lds,
                                            int kdim, int nrows) {
#if __has_builtin(__builtin_amdgcn_tensor_load_to_lds)
  if ((threadIdx.x >> 5) == 0) {
    unsigned long long ga = (unsigned long long)(size_t)gsrc;
    unsigned ldsa = (unsigned)(size_t)lds;          // low 32 bits = LDS offset
    u32x4 g0;
    g0[0] = 1u;                                     // count=1, is_restore=0
    g0[1] = ldsa;                                   // lds_addr
    g0[2] = (unsigned)(ga & 0xFFFFFFFFu);           // global_addr[31:0]
    g0[3] = ((unsigned)(ga >> 32) & 0x01FFFFFFu)    // global_addr[56:32]
            | 0x80000000u;                          // type=2 ("image")
    i32x8 g1;
    g1[0] = (int)(1u << 16);                        // wg_mask=0, data_size=1 (2B)
    g1[1] = (int)((unsigned)(kdim & 0xFFFF) << 16); // tensor_dim0[15:0] @ [63:48]
    g1[2] = (int)(((unsigned)(kdim >> 16) & 0xFFFFu)
            | ((unsigned)(nrows & 0xFFFF) << 16));  // dim0 hi | tensor_dim1 lo
    g1[3] = (int)(((unsigned)(nrows >> 16) & 0xFFFFu)
            | ((unsigned)(kdim & 0xFFFF) << 16));   // dim1 hi | tile_dim0
    g1[4] = (int)((unsigned)(nrows & 0xFFFF));      // tile_dim1 | tile_dim2=0
    g1[5] = kdim;                                   // tensor_dim0_stride[31:0]
    g1[6] = 0;                                      // stride0 hi | stride1 lo
    g1[7] = 0;                                      // stride1 hi
    i32x4 z4 = {0, 0, 0, 0};
    i32x8 z8 = {0, 0, 0, 0, 0, 0, 0, 0};
    __builtin_amdgcn_tensor_load_to_lds(g0, g1, z4, z4, z8, 0);
    __builtin_amdgcn_s_wait_tensorcnt(0);
  }
#else
  int total = nrows * kdim;
  for (int i = threadIdx.x * 8; i < total; i += blockDim.x * 8)
    *(half8*)(lds + i) = *(const half8*)(gsrc + i);
#endif
  __syncthreads();
}

// ---------------------------------------------------------------------------
// Block GEMM mainloop: block tile 128(M) x 64(N), 8 waves, wave = 16x64 strip.
// A streamed from global (register path), B read from LDS panel (ds_load).
// ---------------------------------------------------------------------------
template <int KDIM>
__device__ __forceinline__ void gemm_mainloop(const _Float16* __restrict__ A,
                                              const _Float16* ldsW,
                                              v8f acc[4]) {
  int wave = threadIdx.x >> 5;
  int lane = threadIdx.x & 31;
  const _Float16* Ab = A + ((size_t)blockIdx.x * 128 + wave * 16) * KDIM;
  for (int k = 0; k < KDIM; k += 32) {
    if (k + 64 < KDIM) __builtin_prefetch(Ab + k + 64, 0, 3);
    v16h a = load_A_tile(Ab + k, KDIM, lane);
    #pragma unroll
    for (int j = 0; j < 4; j++) {
      v16h b = load_B_tile(ldsW + (size_t)j * 16 * KDIM + k, KDIM, lane);
      acc[j] = wmma_f16(a, b, acc[j]);
    }
  }
}

// ---------------------------------------------------------------------------
// 0) f32 -> f16 weight cast
// ---------------------------------------------------------------------------
__global__ void cast_f32_f16_kernel(const float* __restrict__ in,
                                    _Float16* __restrict__ out, int n) {
  int i = blockIdx.x * 256 + threadIdx.x;
  if (i < n) out[i] = (_Float16)in[i];
}

// ---------------------------------------------------------------------------
// 1) LayerNorm over channels; x is [B, DMODEL, SEQ]; emit xn_f16 [M, DMODEL]
// ---------------------------------------------------------------------------
__global__ void ln_kernel(const float* __restrict__ x,
                          const float* __restrict__ g,
                          const float* __restrict__ bta,
                          _Float16* __restrict__ xn16) {
  int row = blockIdx.x;                 // b*SEQ + l
  int b   = row >> 11;
  int l   = row & (SEQ - 1);
  const float* xb = x + (size_t)b * DMODEL * SEQ + l;

  __shared__ float red0[8], red1[8];
  float vals[2];
  float s = 0.f, s2 = 0.f;
  #pragma unroll
  for (int i = 0; i < 2; i++) {
    int d = threadIdx.x + i * 256;
    float v = xb[(size_t)d * SEQ];
    vals[i] = v; s += v; s2 += v * v;
  }
  #pragma unroll
  for (int off = 16; off > 0; off >>= 1) {
    s  += __shfl_down(s,  off, 32);
    s2 += __shfl_down(s2, off, 32);
  }
  int wv = threadIdx.x >> 5;
  if ((threadIdx.x & 31) == 0) { red0[wv] = s; red1[wv] = s2; }
  __syncthreads();
  if (threadIdx.x == 0) {
    float a = 0.f, c = 0.f;
    #pragma unroll
    for (int i = 0; i < 8; i++) { a += red0[i]; c += red1[i]; }
    red0[0] = a; red1[0] = c;
  }
  __syncthreads();
  float mean = red0[0] * (1.0f / DMODEL);
  float var  = red1[0] * (1.0f / DMODEL) - mean * mean;
  float inv  = rsqrtf(var + 1e-5f);
  #pragma unroll
  for (int i = 0; i < 2; i++) {
    int d = threadIdx.x + i * 256;
    float v = (vals[i] - mean) * inv * g[d] + bta[d];
    xn16[(size_t)row * DMODEL + d] = (_Float16)v;
  }
}

// ---------------------------------------------------------------------------
// 2) in_proj (TDM): xz[M, 2048] = xn[M,512] @ W[2048,512]^T   (f16 out)
//    grid = (M/128, N/64); weight panel 64x512 f16 = 64 KB LDS via TDM
// ---------------------------------------------------------------------------
__global__ void gemm_in_proj_tdm(const _Float16* __restrict__ A,
                                 const _Float16* __restrict__ W,
                                 _Float16* __restrict__ C) {
  extern __shared__ _Float16 smemW[];
  const int K = DMODEL;
  int n0g = blockIdx.y * 64;
  stage_panel(W + (size_t)n0g * K, smemW, K, 64);

  v8f acc[4] = {};
  gemm_mainloop<DMODEL>(A, smemW, acc);

  int lane = threadIdx.x & 31;
  int wave = threadIdx.x >> 5;
  int m0 = blockIdx.x * 128 + wave * 16 + ((lane & 16) ? 8 : 0);
  #pragma unroll
  for (int j = 0; j < 4; j++) {
    int n = n0g + j * 16 + (lane & 15);
    #pragma unroll
    for (int v = 0; v < 8; v++)
      C[(size_t)(m0 + v) * (2 * DINNER) + n] = (_Float16)acc[j][v];
  }
}

// ---------------------------------------------------------------------------
// 3) depthwise causal conv1d + SiLU : u16[M, DINNER]
// ---------------------------------------------------------------------------
__global__ void conv_silu_kernel(const _Float16* __restrict__ xz16,
                                 const float* __restrict__ cw,
                                 const float* __restrict__ cb,
                                 _Float16* __restrict__ u16) {
  size_t idx = (size_t)blockIdx.x * 256 + threadIdx.x;   // over M*DINNER
  int d = (int)(idx & (DINNER - 1));
  size_t row = idx >> 10;               // b*SEQ + l
  int l = (int)(row & (SEQ - 1));
  float acc = cb[d];
  #pragma unroll
  for (int j = 0; j < DCONV; j++) {
    int ll = l - (DCONV - 1) + j;
    if (ll >= 0) {
      size_t r = row - (size_t)(l - ll);
      acc += cw[d * DCONV + j] * (float)xz16[r * (2 * DINNER) + d];
    }
  }
  float sv = acc / (1.f + __expf(-acc));
  u16[idx] = (_Float16)sv;
}

// ---------------------------------------------------------------------------
// 4) x_proj: x_dbl[M,64] = u[M,1024] @ W[64,1024]^T ; also f16 copy of dt cols
// ---------------------------------------------------------------------------
__global__ void gemm_x_proj(const _Float16* __restrict__ A,
                            const _Float16* __restrict__ W,
                            float* __restrict__ xdbl,
                            _Float16* __restrict__ dt16) {
  int wave = blockIdx.x * (blockDim.x >> 5) + (threadIdx.x >> 5);
  int lane = threadIdx.x & 31;
  const int NT = 64 / 16;               // 4
  int tm = wave / NT, tn = wave % NT;
  const _Float16* Ab = A + (size_t)tm * 16 * DINNER;
  const _Float16* Wb = W + (size_t)tn * 16 * DINNER;
  v8f acc = {};
  for (int k = 0; k < DINNER; k += 32) {
    if (k + 64 < DINNER) {
      __builtin_prefetch(Ab + k + 64, 0, 3);
      __builtin_prefetch(Wb + k + 64, 0, 3);
    }
    v16h a = load_A_tile(Ab + k, DINNER, lane);
    v16h b = load_B_tile(Wb + k, DINNER, lane);
    acc = wmma_f16(a, b, acc);
  }
  int n  = tn * 16 + (lane & 15);
  int m0 = tm * 16 + ((lane & 16) ? 8 : 0);
  #pragma unroll
  for (int v = 0; v < 8; v++) {
    float val = acc[v];
    xdbl[(size_t)(m0 + v) * 64 + n] = val;
    if (tn < 2)                                  // cols 0..31 -> dt rank part
      dt16[(size_t)(m0 + v) * DTRANK + n] = (_Float16)val;
  }
}

// ---------------------------------------------------------------------------
// 5) dt_proj: dt[M,1024] = softplus(dt_r[M,32] @ W[1024,32]^T + b)  (K==32)
// ---------------------------------------------------------------------------
__global__ void gemm_dt_proj(const _Float16* __restrict__ A,
                             const _Float16* __restrict__ W,
                             const float* __restrict__ bias,
                             float* __restrict__ dtf) {
  int wave = blockIdx.x * (blockDim.x >> 5) + (threadIdx.x >> 5);
  int lane = threadIdx.x & 31;
  const int NT = DINNER / 16;           // 64
  int tm = wave / NT, tn = wave % NT;
  const _Float16* Ab = A + (size_t)tm * 16 * DTRANK;
  const _Float16* Wb = W + (size_t)tn * 16 * DTRANK;
  v8f acc = {};
  v16h a = load_A_tile(Ab, DTRANK, lane);
  v16h b = load_B_tile(Wb, DTRANK, lane);
  acc = wmma_f16(a, b, acc);
  int n  = tn * 16 + (lane & 15);
  int m0 = tm * 16 + ((lane & 16) ? 8 : 0);
  #pragma unroll
  for (int v = 0; v < 8; v++) {
    float val = acc[v] + bias[n];
    if (val < 20.f) val = log1pf(__expf(val));   // softplus
    dtf[(size_t)(m0 + v) * DINNER + n] = val;
  }
}

// ---------------------------------------------------------------------------
// 6) SSM scan. One lane per (b, d) channel, 16 states in VGPRs.
// ---------------------------------------------------------------------------
#define SCAN_CHUNK 64
__global__ void scan_kernel(const float* __restrict__ dtf,
                            const _Float16* __restrict__ u16,
                            const _Float16* __restrict__ xz16,
                            const float* __restrict__ xdbl,
                            const float* __restrict__ A_log,
                            const float* __restrict__ Dvec,
                            _Float16* __restrict__ y16) {
  __shared__ float bc[SCAN_CHUNK * 2 * DSTATE];          // 8 KB
  const int tid = threadIdx.x;                           // 0..127
  const int b   = blockIdx.x >> 3;
  const int d   = (blockIdx.x & 7) * 128 + tid;

  float Ad[DSTATE], h[DSTATE];
  #pragma unroll
  for (int s = 0; s < DSTATE; s++) {
    Ad[s] = -__expf(A_log[d * DSTATE + s]);
    h[s]  = 0.f;
  }
  const float Dd = Dvec[d];

  for (int t0 = 0; t0 < SEQ; t0 += SCAN_CHUNK) {
    #pragma unroll
    for (int i = 0; i < (SCAN_CHUNK * 2 * DSTATE) / 128; i++) {
      int idx = tid + i * 128;
      int tt  = idx >> 5;
      int c   = idx & 31;
      bc[idx] = xdbl[((size_t)b * SEQ + t0 + tt) * 64 + DTRANK + c];
    }
    __syncthreads();
    for (int tt = 0; tt < SCAN_CHUNK; tt++) {
      size_t row = (size_t)b * SEQ + t0 + tt;
      float dt = dtf[row * DINNER + d];
      float u  = (float)u16[row * DINNER + d];
      float du = dt * u;
      float y  = 0.f;
      const float* Bp = &bc[tt * 2 * DSTATE];
      const float* Cp = Bp + DSTATE;
      #pragma unroll
      for (int s = 0; s < DSTATE; s++) {
        float dA = __expf(dt * Ad[s]);
        h[s] = dA * h[s] + du * Bp[s];
        y   += h[s] * Cp[s];
      }
      y += Dd * u;
      float z = (float)xz16[row * (2 * DINNER) + DINNER + d];
      y *= z / (1.f + __expf(-z));
      y16[row * DINNER + d] = (_Float16)y;
    }
    __syncthreads();
  }
}

// ---------------------------------------------------------------------------
// 7) out_proj (TDM) fused with residual + [B,L,D]->[B,D,L] transpose
//    grid = (M/128, DMODEL/64); weight panel 64x1024 f16 = 128 KB LDS via TDM
// ---------------------------------------------------------------------------
__global__ void gemm_out_proj_tdm(const _Float16* __restrict__ A,
                                  const _Float16* __restrict__ W,
                                  const float* __restrict__ xres,
                                  float* __restrict__ out) {
  extern __shared__ _Float16 smemW[];
  const int K = DINNER;
  int n0g = blockIdx.y * 64;
  stage_panel(W + (size_t)n0g * K, smemW, K, 64);

  v8f acc[4] = {};
  gemm_mainloop<DINNER>(A, smemW, acc);

  int lane = threadIdx.x & 31;
  int wave = threadIdx.x >> 5;
  int m0 = blockIdx.x * 128 + wave * 16 + ((lane & 16) ? 8 : 0);
  #pragma unroll
  for (int j = 0; j < 4; j++) {
    int n = n0g + j * 16 + (lane & 15);
    #pragma unroll
    for (int v = 0; v < 8; v++) {
      int row = m0 + v;                 // b*SEQ + l ; tile never crosses b
      int b = row >> 11, l = row & (SEQ - 1);
      size_t oi = ((size_t)b * DMODEL + n) * SEQ + l;
      out[oi] = acc[j][v] + xres[oi];
    }
  }
}

// ---------------------------------------------------------------------------
// launch
// ---------------------------------------------------------------------------
extern "C" void kernel_launch(void* const* d_in, const int* in_sizes, int n_in,
                              void* d_out, int out_size, void* d_ws, size_t ws_size,
                              hipStream_t stream) {
  (void)in_sizes; (void)n_in; (void)out_size; (void)ws_size;
  const float* x      = (const float*)d_in[0];
  const float* ln_g   = (const float*)d_in[1];
  const float* ln_b   = (const float*)d_in[2];
  const float* in_w   = (const float*)d_in[3];
  const float* conv_w = (const float*)d_in[4];
  const float* conv_b = (const float*)d_in[5];
  const float* xp_w   = (const float*)d_in[6];
  const float* dt_w   = (const float*)d_in[7];
  const float* dt_b   = (const float*)d_in[8];
  const float* A_log  = (const float*)d_in[9];
  const float* Dvec   = (const float*)d_in[10];
  const float* out_w  = (const float*)d_in[11];
  float* out = (float*)d_out;

  char* ws = (char*)d_ws;
  size_t o = 0;
  auto alloc = [&](size_t bytes) {
    size_t r = o; o += (bytes + 255) & ~(size_t)255; return r;
  };
  _Float16* xn16   = (_Float16*)(ws + alloc((size_t)MROWS * DMODEL * 2));
  _Float16* win16  = (_Float16*)(ws + alloc((size_t)2 * DINNER * DMODEL * 2));
  _Float16* wxp16  = (_Float16*)(ws + alloc((size_t)64 * DINNER * 2));
  _Float16* wdt16  = (_Float16*)(ws + alloc((size_t)DINNER * DTRANK * 2));
  _Float16* wout16 = (_Float16*)(ws + alloc((size_t)DMODEL * DINNER * 2));
  _Float16* xz16   = (_Float16*)(ws + alloc((size_t)MROWS * 2 * DINNER * 2));
  _Float16* u16    = (_Float16*)(ws + alloc((size_t)MROWS * DINNER * 2));
  float*    xdbl   = (float*)   (ws + alloc((size_t)MROWS * 64 * 4));
  _Float16* dt16   = (_Float16*)(ws + alloc((size_t)MROWS * DTRANK * 2));
  float*    dtf    = (float*)   (ws + alloc((size_t)MROWS * DINNER * 4));
  _Float16* y16    = (_Float16*)(ws + alloc((size_t)MROWS * DINNER * 2));

  {
    int n1 = 2 * DINNER * DMODEL, n2 = 64 * DINNER,
        n3 = DINNER * DTRANK,     n4 = DMODEL * DINNER;
    cast_f32_f16_kernel<<<(n1 + 255) / 256, 256, 0, stream>>>(in_w,  win16,  n1);
    cast_f32_f16_kernel<<<(n2 + 255) / 256, 256, 0, stream>>>(xp_w,  wxp16,  n2);
    cast_f32_f16_kernel<<<(n3 + 255) / 256, 256, 0, stream>>>(dt_w,  wdt16,  n3);
    cast_f32_f16_kernel<<<(n4 + 255) / 256, 256, 0, stream>>>(out_w, wout16, n4);
  }
  // 1) layernorm
  ln_kernel<<<MROWS, 256, 0, stream>>>(x, ln_g, ln_b, xn16);
  // 2) in_proj: block tile 128x64, TDM-staged weight panel (64 KB dynamic LDS)
  gemm_in_proj_tdm<<<dim3(MROWS / 128, (2 * DINNER) / 64), 256,
                     64 * DMODEL * 2, stream>>>(xn16, win16, xz16);
  // 3) conv + SiLU
  conv_silu_kernel<<<(MROWS * DINNER) / 256, 256, 0, stream>>>(xz16, conv_w, conv_b, u16);
  // 4) x_proj: 512*4 = 2048 wave-tiles, direct path
  gemm_x_proj<<<2048 / 8, 256, 0, stream>>>(u16, wxp16, xdbl, dt16);
  // 5) dt_proj: 512*64 = 32768 wave-tiles, direct path
  gemm_dt_proj<<<32768 / 8, 256, 0, stream>>>(dt16, wdt16, dt_b, dtf);
  // 6) scan
  scan_kernel<<<BATCH * 8, 128, 0, stream>>>(dtf, u16, xz16, xdbl, A_log, Dvec, y16);
  // 7) out_proj: TDM-staged panel (128 KB dynamic LDS), fused residual+transpose
  gemm_out_proj_tdm<<<dim3(MROWS / 128, DMODEL / 64), 256,
                      64 * DINNER * 2, stream>>>(y16, wout16, x, out);
}
